// HGT_49692771615035
// MI455X (gfx1250) — compile-verified, matrix operand order checked by reading, not run on hardware
//
#include <hip/hip_runtime.h>
#include <math.h>

// ---------------------------------------------------------------------------
// HGT forward for MI455X (gfx1250, wave32).
// All dense math runs through v_wmma_f32_16x16x32_bf16 (bf16 in, fp32 acc).
// Edge attention relation transforms are hoisted from per-edge to per-source-
// node batched GEMMs (4x less compute, WMMA-friendly). Softmax/LN/reductions
// stay fp32/fp64. Workspace demand ~0.8 GB.
// Round 2 change: use native fptrunc (v_cvt_pk_bf16_f32) instead of manual
// integer RNE for fp32->bf16, removing ~50 VALU ops per GEMM k-step.
// ---------------------------------------------------------------------------

typedef __attribute__((ext_vector_type(16))) __bf16 v16bf;
typedef __attribute__((ext_vector_type(8)))  float  v8f;

#define NOPN 65536
#define NVARN 65536
#define NDIRN 2048
#define NFN 16384
#define ECD 262144
#define EDIR 32768
#define CDIM 256
#define HN 4
#define DN 64
#define BN 16

__device__ __forceinline__ float geluf(float x) {
    return 0.5f * x * (1.0f + erff(x * 0.70710678118654752f));
}

__device__ __forceinline__ __bf16 f2bf(float f) {
    return (__bf16)f;   // lowers to v_cvt_pk_bf16_f32 (paired) on gfx1250
}

__device__ __forceinline__ void atomicMaxF(float* addr, float v) {
    if (v >= 0.f) atomicMax((int*)addr, __float_as_int(v));
    else          atomicMin((unsigned int*)addr, __float_as_uint(v));
}

// --------------------------- WMMA fragment loads ---------------------------
// A (16x32 bf16, MxK): lane m = lane&15, half = lane>>4.
// reg r<4  : K = 2r + 8*half  (pair K,K+1)
// reg r>=4 : K = 2(r-4)+16+8*half
__device__ __forceinline__ int frag_k(int r, int hi) {
    return ((r < 4) ? (2 * r) : (2 * (r - 4) + 16)) + hi * 8;
}

__device__ __forceinline__ v16bf load_a_frag(const float* Arow, int kk, int hi, int aGelu) {
    v16bf a;
#pragma unroll
    for (int r = 0; r < 8; ++r) {
        int k = frag_k(r, hi);
        float f0 = Arow[kk + k];
        float f1 = Arow[kk + k + 1];
        if (aGelu) { f0 = geluf(f0); f1 = geluf(f1); }
        a[2 * r] = f2bf(f0);
        a[2 * r + 1] = f2bf(f1);
    }
    return a;
}

// B stored transposed [N,K] bf16; lane n = lane&15 selects row; K pairs contiguous.
__device__ __forceinline__ v16bf load_b_frag(const __bf16* Bt, long row, int ldb, int kk, int hi) {
    const __bf16* p = Bt + row * (long)ldb + kk;
    v16bf b;
#pragma unroll
    for (int r = 0; r < 8; ++r) {
        int k = frag_k(r, hi);
        b[2 * r] = p[k];
        b[2 * r + 1] = p[k + 1];
    }
    return b;
}

__device__ __forceinline__ void epilogue_tile(v8f acc, int n, int row0, int hi,
                                              const float* bias, float* Cb, int ldc,
                                              const int* riC, const float* Res, int ldr,
                                              int mode, float sgate) {
    float bv = bias ? bias[n] : 0.f;
#pragma unroll
    for (int r = 0; r < 8; ++r) {
        int m = row0 + r + hi * 8;
        long rowC = riC ? (long)riC[m] : (long)m;
        float v = acc[r] + bv;
        if (mode == 1) v = geluf(v);
        else if (mode == 2) v = sgate * v + (1.f - sgate) * Res[rowC * (long)ldr + n];
        Cb[rowC * (long)ldc + n] = v;
    }
}

// Generic GEMM: C[M,N] = epilogue(A[M,K] @ Bt[N,K]^T + bias)
//   block = 128 threads (4 waves); wave w -> 16-row tile, 64-col strip (4 WMMA tiles)
//   grid  = (M/64, N/64, batch); strides sA/sB/sC are per-batch ELEMENT offsets.
//   riA gathers A rows, riC scatters C rows (and Res rows). modes: 0 linear,
//   1 GELU, 2 sigmoid(skip)-gated residual blend.
__global__ void gemm_wmma_bf16(const float* __restrict__ A, long sA, int lda,
                               const int* __restrict__ riA, int aGelu,
                               const __bf16* __restrict__ Bt, long sB, int ldb,
                               const float* __restrict__ bias,
                               float* __restrict__ C, long sC, int ldc,
                               const int* __restrict__ riC,
                               const float* __restrict__ Res, int ldr,
                               const float* __restrict__ skipPtr,
                               int M, int N, int K, int mode) {
    int w = threadIdx.x >> 5;
    int lane = threadIdx.x & 31;
    int mTile = blockIdx.x * 4 + w;
    int row0 = mTile * 16;
    if (row0 >= M) return;
    int col0 = blockIdx.y * 64;
    int b = blockIdx.z;

    const float* Ab = A + (long)b * sA;
    const __bf16* Bb = Bt + (long)b * sB;
    float* Cb = C + (long)b * sC;

    int hi = lane >> 4;
    int nB = lane & 15;
    int mA = row0 + (lane & 15);
    long rowA = riA ? (long)riA[mA] : (long)mA;
    const float* Arow = Ab + rowA * (long)lda;

    v8f acc0 = {}, acc1 = {}, acc2 = {}, acc3 = {};
    for (int kk = 0; kk < K; kk += 32) {
        if (kk + 32 < K) __builtin_prefetch(Arow + kk + 32, 0, 1);  // global_prefetch_b8
        v16bf a = load_a_frag(Arow, kk, hi, aGelu);
        v16bf b0 = load_b_frag(Bb, col0 + 0 * 16 + nB, ldb, kk, hi);
        v16bf b1 = load_b_frag(Bb, col0 + 1 * 16 + nB, ldb, kk, hi);
        v16bf b2 = load_b_frag(Bb, col0 + 2 * 16 + nB, ldb, kk, hi);
        v16bf b3 = load_b_frag(Bb, col0 + 3 * 16 + nB, ldb, kk, hi);
        acc0 = __builtin_amdgcn_wmma_f32_16x16x32_bf16(false, a, false, b0, (short)0, acc0, false, false);
        acc1 = __builtin_amdgcn_wmma_f32_16x16x32_bf16(false, a, false, b1, (short)0, acc1, false, false);
        acc2 = __builtin_amdgcn_wmma_f32_16x16x32_bf16(false, a, false, b2, (short)0, acc2, false, false);
        acc3 = __builtin_amdgcn_wmma_f32_16x16x32_bf16(false, a, false, b3, (short)0, acc3, false, false);
    }
    float sgate = 0.f;
    if (mode == 2) sgate = 1.f / (1.f + __expf(-*skipPtr));
    epilogue_tile(acc0, col0 + 0 * 16 + nB, row0, hi, bias, Cb, ldc, riC, Res, ldr, mode, sgate);
    epilogue_tile(acc1, col0 + 1 * 16 + nB, row0, hi, bias, Cb, ldc, riC, Res, ldr, mode, sgate);
    epilogue_tile(acc2, col0 + 2 * 16 + nB, row0, hi, bias, Cb, ldc, riC, Res, ldr, mode, sgate);
    epilogue_tile(acc3, col0 + 3 * 16 + nB, row0, hi, bias, Cb, ldc, riC, Res, ldr, mode, sgate);
}

// ------------------------------ small kernels ------------------------------

__global__ void fill_f32(float* p, long n, float v) {
    long i = (long)blockIdx.x * blockDim.x + threadIdx.x;
    if (i < n) p[i] = v;
}

// fp32 [K,N] row-major -> bf16 [N,K] (per batch matrix)
__global__ void transpose_to_bf16(const float* __restrict__ src, __bf16* __restrict__ dst,
                                  int K, int N) {
    long base = (long)blockIdx.z * K * N;
    int idx = blockIdx.x * blockDim.x + threadIdx.x;
    if (idx >= K * N) return;
    int k = idx / N, n = idx % N;
    dst[base + (long)n * K + k] = f2bf(src[base + idx]);
}

__global__ void reduce_sumsq(const float* __restrict__ x, long n, double* out) {
    __shared__ double sh0[256], sh1[256];
    long i = (long)blockIdx.x * blockDim.x + threadIdx.x;
    long stride = (long)gridDim.x * blockDim.x;
    double s = 0.0, s2 = 0.0;
    for (; i < n; i += stride) { double v = x[i]; s += v; s2 += v * v; }
    sh0[threadIdx.x] = s; sh1[threadIdx.x] = s2;
    __syncthreads();
    for (int o = 128; o > 0; o >>= 1) {
        if ((int)threadIdx.x < o) { sh0[threadIdx.x] += sh0[threadIdx.x + o]; sh1[threadIdx.x] += sh1[threadIdx.x + o]; }
        __syncthreads();
    }
    if (threadIdx.x == 0) { atomicAdd(&out[0], sh0[0]); atomicAdd(&out[1], sh1[0]); }
}

// graph-mode LayerNorm (mean/var over whole [N,C] tensor), in place
__global__ void ln_apply(float* x, long NC, int Cc, const double* st,
                         const float* __restrict__ w, const float* __restrict__ b) {
    long i = (long)blockIdx.x * blockDim.x + threadIdx.x;
    if (i >= NC) return;
    double cnt = (double)NC;
    double mean = st[0] / cnt;
    double var = st[1] / cnt - mean * mean;
    float sc = rsqrtf((float)var + 1e-5f);
    int c = (int)(i % Cc);
    x[i] = ((x[i] - (float)mean) * sc) * w[c] + b[c];
}

// a[e,h] = dot(Q[dst[e], h*64:...], Kp[src[e], h*64:...]) * pr[h] / 8
__global__ void edge_alpha(const float* __restrict__ Q, int ldq,
                           const float* __restrict__ Kp,
                           const int* __restrict__ src, const int* __restrict__ dst,
                           int E, const float* __restrict__ pr, float* __restrict__ alpha) {
    int idx = blockIdx.x * blockDim.x + threadIdx.x;
    if (idx >= E * HN) return;
    int e = idx >> 2, h = idx & 3;
    const float4* q4 = (const float4*)(Q + (long)dst[e] * ldq + h * 64);
    const float4* k4 = (const float4*)(Kp + (long)src[e] * CDIM + h * 64);
    float acc = 0.f;
#pragma unroll
    for (int d = 0; d < 16; ++d) {
        float4 a = q4[d], b = k4[d];
        acc += a.x * b.x + a.y * b.y + a.z * b.z + a.w * b.w;
    }
    alpha[idx] = acc * pr[h] * 0.125f;  // 1/sqrt(64)
}

__global__ void seg_max_k(const float* __restrict__ alpha, const int* __restrict__ dst,
                          int E, float* m) {
    int idx = blockIdx.x * blockDim.x + threadIdx.x;
    if (idx >= E * HN) return;
    int e = idx >> 2, h = idx & 3;
    atomicMaxF(&m[(long)dst[e] * HN + h], alpha[idx]);
}

__global__ void seg_expsum_k(float* alpha, const int* __restrict__ dst, int E,
                             const float* __restrict__ m, float* s) {
    int idx = blockIdx.x * blockDim.x + threadIdx.x;
    if (idx >= E * HN) return;
    int e = idx >> 2, h = idx & 3;
    float ex = expf(alpha[idx] - m[(long)dst[e] * HN + h]);
    alpha[idx] = ex;
    atomicAdd(&s[(long)dst[e] * HN + h], ex);
}

__global__ void seg_agg(const float* __restrict__ alpha, const int* __restrict__ src,
                        const int* __restrict__ dst, int E, const float* __restrict__ s,
                        const float* __restrict__ Vp, float* agg) {
    int idx = blockIdx.x * blockDim.x + threadIdx.x;
    if (idx >= E * HN) return;
    int e = idx >> 2, h = idx & 3;
    int d0 = dst[e];
    float w = alpha[idx] / (s[(long)d0 * HN + h] + 1e-16f);
    const float4* v4 = (const float4*)(Vp + (long)src[e] * CDIM + h * 64);
    float* ag = agg + (long)d0 * CDIM + h * 64;
#pragma unroll
    for (int d = 0; d < 16; ++d) {
        float4 v = v4[d];
        atomicAdd(ag + 4 * d + 0, w * v.x);
        atomicAdd(ag + 4 * d + 1, w * v.y);
        atomicAdd(ag + 4 * d + 2, w * v.z);
        atomicAdd(ag + 4 * d + 3, w * v.w);
    }
}

// segment add+max pooling; batch is sorted so we accumulate runs before atomics
__global__ void pool_accum(const float* __restrict__ x, const int* __restrict__ batch,
                           int N, float* padd, float* pmax) {
    int c = threadIdx.x;                       // 256 channels
    int n0 = blockIdx.x * 256;
    int n1 = n0 + 256; if (n1 > N) n1 = N;
    int curb = -1; float sum = 0.f, mx = -__builtin_huge_valf();
    for (int n = n0; n < n1; ++n) {
        int b = batch[n];
        if (b != curb) {
            if (curb >= 0) { atomicAdd(&padd[curb * 256 + c], sum); atomicMaxF(&pmax[curb * 256 + c], mx); }
            curb = b; sum = 0.f; mx = -__builtin_huge_valf();
        }
        float v = x[(long)n * 256 + c];
        sum += v; mx = fmaxf(mx, v);
    }
    if (curb >= 0) { atomicAdd(&padd[curb * 256 + c], sum); atomicMaxF(&pmax[curb * 256 + c], mx); }
}

__global__ void yb_mlp(const float* yb_in, const float* W1, const float* b1,
                       const float* W2, const float* b2, float* yb) {
    __shared__ float h[16][16];
    int i = threadIdx.x >> 4, j = threadIdx.x & 15;
    float t = yb_in[i] * W1[j] + b1[j];
    h[i][j] = t > 0.f ? t : 0.2f * t;  // leaky_relu(0.2)
    __syncthreads();
    float acc = b2[j];
    for (int k = 0; k < 16; ++k) acc += h[i][k] * W2[k * 16 + j];
    yb[i * 16 + j] = acc;
}

__global__ void assemble_g(const float* pao, const float* pmo, const float* pav,
                           const float* pmv, const float* yb, float* g) {
    int idx = blockIdx.x * blockDim.x + threadIdx.x;
    if (idx >= 16 * 1040) return;
    int i = idx / 1040, j = idx % 1040;
    float v;
    if (j < 256) v = pao[i * 256 + j];
    else if (j < 512) v = pmo[i * 256 + j - 256];
    else if (j < 768) v = pav[i * 256 + j - 512];
    else if (j < 1024) v = pmv[i * 256 + j - 768];
    else v = yb[i * 16 + (j - 1024)];
    g[idx] = v;
}

__global__ void final_mlp(const float* __restrict__ g,
                          const float* W1, const float* b1,
                          const float* g1, const float* bb1, const float* m1, const float* v1,
                          const float* W2, const float* b2,
                          const float* g2, const float* bb2, const float* m2, const float* v2,
                          const float* W3, const float* b3, float* out) {
    __shared__ float h1[256];
    __shared__ float h2[64];
    int i = blockIdx.x, t = threadIdx.x;
    float acc = b1[t];
    for (int k = 0; k < 1040; ++k) acc += g[i * 1040 + k] * W1[k * 256 + t];
    acc = (acc - m1[t]) * rsqrtf(v1[t] + 1e-5f) * g1[t] + bb1[t];
    h1[t] = geluf(acc);
    __syncthreads();
    if (t < 64) {
        float a2 = b2[t];
        for (int k = 0; k < 256; ++k) a2 += h1[k] * W2[k * 64 + t];
        a2 = (a2 - m2[t]) * rsqrtf(v2[t] + 1e-5f) * g2[t] + bb2[t];
        h2[t] = geluf(a2);
    }
    __syncthreads();
    if (t == 0) {
        float a3 = b3[0];
        for (int k = 0; k < 64; ++k) a3 += h2[k] * W3[k];
        out[i] = a3;
    }
}

// --------------------------------- driver ----------------------------------

extern "C" void kernel_launch(void* const* d_in, const int* in_sizes, int n_in,
                              void* d_out, int out_size, void* d_ws, size_t ws_size,
                              hipStream_t stream) {
    (void)in_sizes; (void)n_in; (void)out_size; (void)ws_size;
    auto F = [&](int i) { return (const float*)d_in[i]; };
    auto I = [&](int i) { return (const int*)d_in[i]; };

    const float *x_op_in = F(0), *x_var_in = F(1), *x_dir_in = F(2);
    const int *e_oo = I(3), *e_ov = I(4), *e_vo = I(5);
    const int *op_fidx = I(6), *var_fidx = I(7);
    const int *e_do_s = I(8), *e_do_d = I(9), *e_dv_s = I(10), *e_dv_d = I(11);
    const int *batch_op = I(12), *batch_var = I(13);
    const float* y_base = F(14);
    const float *proj_W = F(15), *proj_b = F(16);
    const float *hls_kqv_W = F(17), *hls_kqv_b = F(18);
    const float *hls_kr = F(19), *hls_vr = F(20), *hls_pr = F(21);
    const float *hls_out_W = F(22), *hls_out_b = F(23), *hls_skip = F(24);
    const float *norm_w = F(25), *norm_b = F(26);
    const float *conv_kqv_W = F(27), *conv_kqv_b = F(28);
    const float *conv_kr = F(29), *conv_vr = F(30), *conv_pr = F(31);
    const float *conv_out_W = F(32), *conv_out_b = F(33), *conv_skip = F(34);
    const float *yb_W1 = F(35), *yb_b1 = F(36), *yb_W2 = F(37), *yb_b2 = F(38);
    const float *g_W1 = F(39), *g_b1 = F(40);
    const float *bn1_g = F(41), *bn1_b = F(42), *bn1_rm = F(43), *bn1_rv = F(44);
    const float *g_W2 = F(45), *g_b2 = F(46);
    const float *bn2_g = F(47), *bn2_b = F(48), *bn2_rm = F(49), *bn2_rv = F(50);
    const float *g_W3 = F(51), *g_b3 = F(52);

    // -------- workspace carve-out --------
    char* wsB = (char*)d_ws;
    size_t off = 0;
    auto allocB = [&](size_t bytes) -> char* {
        size_t a = (off + 255) & ~(size_t)255;
        off = a + bytes;
        return wsB + a;
    };
    auto allocF = [&](size_t n) { return (float*)allocB(n * 4); };
    auto allocH = [&](size_t n) { return (__bf16*)allocB(n * 2); };

    float* x_op = allocF((size_t)NOPN * CDIM);
    float* x_var = allocF((size_t)NVARN * CDIM);
    float* x_dir = allocF((size_t)NDIRN * CDIM);
    float* z_op = allocF((size_t)NOPN * 768);
    float* z_var = allocF((size_t)NVARN * 768);
    float* z_dir = allocF((size_t)NDIRN * 768);
    float* q_fop = allocF((size_t)NFN * CDIM);
    float* q_fvar = allocF((size_t)NFN * CDIM);
    float* Rp = allocF((size_t)NOPN * CDIM);            // shared K'/V' scratch
    float* alphab[3] = { allocF((size_t)ECD * HN), allocF((size_t)ECD * HN), allocF((size_t)ECD * HN) };
    float* m_op = allocF((size_t)NOPN * HN);
    float* s_op = allocF((size_t)NOPN * HN);
    float* m_var = allocF((size_t)NVARN * HN);
    float* s_var = allocF((size_t)NVARN * HN);
    float* agg_op = allocF((size_t)NOPN * CDIM);
    float* agg_var = allocF((size_t)NVARN * CDIM);
    double* stats = (double*)allocB(16);
    float* padd_op = allocF(16 * CDIM);
    float* pmax_op = allocF(16 * CDIM);
    float* padd_var = allocF(16 * CDIM);
    float* pmax_var = allocF(16 * CDIM);
    float* ybuf = allocF(16 * 16);
    float* gbuf = allocF(16 * 1040);

    __bf16* projT    = allocH((size_t)3 * 128 * CDIM);
    __bf16* hlskqvT  = allocH((size_t)3 * 768 * CDIM);
    __bf16* hlskrT   = allocH((size_t)8 * DN * DN);
    __bf16* hlsvrT   = allocH((size_t)8 * DN * DN);
    __bf16* hlsoutT  = allocH((size_t)2 * CDIM * CDIM);
    __bf16* convkqvT = allocH((size_t)8 * 768 * CDIM);
    __bf16* convkrT  = allocH((size_t)48 * DN * DN);
    __bf16* convvrT  = allocH((size_t)48 * DN * DN);
    __bf16* convoutT = allocH((size_t)8 * CDIM * CDIM);

    const float NEG_INF = -__builtin_huge_valf();
    auto fill = [&](float* p, long n, float v) {
        fill_f32<<<dim3((unsigned)((n + 255) / 256)), 256, 0, stream>>>(p, n, v);
    };
    auto tr = [&](const float* src, __bf16* dst, int K, int N, int batch) {
        transpose_to_bf16<<<dim3((unsigned)((K * N + 255) / 256), 1, (unsigned)batch),
                            256, 0, stream>>>(src, dst, K, N);
    };
    auto gemm = [&](const float* A, long sA, int lda, const int* riA, int aGelu,
                    const __bf16* Bt, long sB, int ldb, const float* bias,
                    float* Cm, long sC, int ldc, const int* riC,
                    const float* Res, const float* skipPtr,
                    int M, int N, int K, int mode, int batch) {
        dim3 grd((unsigned)((M + 63) / 64), (unsigned)(N / 64), (unsigned)batch);
        gemm_wmma_bf16<<<grd, 128, 0, stream>>>(A, sA, lda, riA, aGelu, Bt, sB, ldb, bias,
                                                Cm, sC, ldc, riC, Res, ldc, skipPtr,
                                                M, N, K, mode);
    };
    auto eblk = [&](int E) { return dim3((unsigned)((E * HN + 255) / 256)); };

    // -------- weight transposes (fp32 [K,N] -> bf16 [N,K]) --------
    tr(proj_W, projT, 128, CDIM, 3);
    tr(hls_kqv_W, hlskqvT, CDIM, 768, 3);
    tr(hls_kr, hlskrT, DN, DN, 8);
    tr(hls_vr, hlsvrT, DN, DN, 8);
    tr(hls_out_W, hlsoutT, CDIM, CDIM, 2);
    tr(conv_kqv_W, convkqvT, CDIM, 768, 8);
    tr(conv_kr, convkrT, DN, DN, 48);
    tr(conv_vr, convvrT, DN, DN, 48);
    tr(conv_out_W, convoutT, CDIM, CDIM, 8);

    // -------- proj_in + GELU --------
    gemm(x_op_in, 0, 128, nullptr, 0, projT + 0 * 128 * CDIM, 0, 128, proj_b + 0 * CDIM,
         x_op, 0, CDIM, nullptr, nullptr, nullptr, NOPN, CDIM, 128, 1, 1);
    gemm(x_var_in, 0, 128, nullptr, 0, projT + 1 * (size_t)128 * CDIM, 0, 128, proj_b + 1 * CDIM,
         x_var, 0, CDIM, nullptr, nullptr, nullptr, NVARN, CDIM, 128, 1, 1);
    gemm(x_dir_in, 0, 128, nullptr, 0, projT + 2 * (size_t)128 * CDIM, 0, 128, proj_b + 2 * CDIM,
         x_dir, 0, CDIM, nullptr, nullptr, nullptr, NDIRN, CDIM, 128, 1, 1);

    // -------- HLS directive stage (filtered subgraph) --------
    // q for filtered op/var nodes (q slice = rows 256..511 of transposed kqv W)
    gemm(x_op, 0, CDIM, op_fidx, 0, hlskqvT + (size_t)0 * 768 * CDIM + 256 * CDIM, 0, CDIM,
         hls_kqv_b + 0 * 768 + 256, q_fop, 0, CDIM, nullptr, nullptr, nullptr,
         NFN, CDIM, CDIM, 0, 1);
    gemm(x_var, 0, CDIM, var_fidx, 0, hlskqvT + (size_t)1 * 768 * CDIM + 256 * CDIM, 0, CDIM,
         hls_kqv_b + 1 * 768 + 256, q_fvar, 0, CDIM, nullptr, nullptr, nullptr,
         NFN, CDIM, CDIM, 0, 1);
    // full kqv for dir nodes
    gemm(x_dir, 0, CDIM, nullptr, 0, hlskqvT + (size_t)2 * 768 * CDIM, 0, CDIM,
         hls_kqv_b + 2 * 768, z_dir, 0, 768, nullptr, nullptr, nullptr,
         NDIRN, 768, CDIM, 0, 1);

    for (int t = 0; t < 2; ++t) {
        const int* es = (t == 0) ? e_do_s : e_dv_s;
        const int* ed = (t == 0) ? e_do_d : e_dv_d;
        const float* qn = (t == 0) ? q_fop : q_fvar;
        float* mB = (t == 0) ? m_op : m_var;
        float* sB = (t == 0) ? s_op : s_var;
        float* aggB = (t == 0) ? agg_op : agg_var;
        fill(mB, (long)NFN * HN, NEG_INF);
        fill(sB, (long)NFN * HN, 0.f);
        fill(aggB, (long)NFN * CDIM, 0.f);
        // K' = k_dir @ kr[t]  (batched over heads)
        gemm(z_dir + 0, 64, 768, nullptr, 0, hlskrT + (size_t)t * HN * DN * DN, DN * DN, DN,
             nullptr, Rp, 64, CDIM, nullptr, nullptr, nullptr, NDIRN, DN, DN, 0, HN);
        edge_alpha<<<eblk(EDIR), 256, 0, stream>>>(qn, CDIM, Rp, es, ed, EDIR,
                                                   hls_pr + t * HN, alphab[0]);
        seg_max_k<<<eblk(EDIR), 256, 0, stream>>>(alphab[0], ed, EDIR, mB);
        seg_expsum_k<<<eblk(EDIR), 256, 0, stream>>>(alphab[0], ed, EDIR, mB, sB);
        // V' = v_dir @ vr[t]
        gemm(z_dir + 512, 64, 768, nullptr, 0, hlsvrT + (size_t)t * HN * DN * DN, DN * DN, DN,
             nullptr, Rp, 64, CDIM, nullptr, nullptr, nullptr, NDIRN, DN, DN, 0, HN);
        seg_agg<<<eblk(EDIR), 256, 0, stream>>>(alphab[0], es, ed, EDIR, sB, Rp, aggB);
    }
    // node_out with scatter back through fidx (skip-gated residual, in place)
    gemm(agg_op, 0, CDIM, nullptr, 1, hlsoutT + 0, 0, CDIM, hls_out_b + 0,
         x_op, 0, CDIM, op_fidx, x_op, hls_skip + 0, NFN, CDIM, CDIM, 2, 1);
    gemm(agg_var, 0, CDIM, nullptr, 1, hlsoutT + (size_t)CDIM * CDIM, 0, CDIM, hls_out_b + CDIM,
         x_var, 0, CDIM, var_fidx, x_var, hls_skip + 1, NFN, CDIM, CDIM, 2, 1);

    // -------- 4 x (graph-LN + HGTConv on CDFG) --------
    const int* srcs[3] = { e_oo, e_ov, e_vo };
    const int* dsts[3] = { e_oo + ECD, e_ov + ECD, e_vo + ECD };
    const long NC = (long)NOPN * CDIM;
    dim3 lnGrid((unsigned)((NC + 255) / 256));

    for (int L = 0; L < 4; ++L) {
        // LayerNorm (graph mode), in place on x
        fill((float*)stats, 4, 0.f);
        reduce_sumsq<<<1024, 256, 0, stream>>>(x_op, NC, stats);
        ln_apply<<<lnGrid, 256, 0, stream>>>(x_op, NC, CDIM, stats,
                                             norm_w + (L * 2 + 0) * CDIM, norm_b + (L * 2 + 0) * CDIM);
        fill((float*)stats, 4, 0.f);
        reduce_sumsq<<<1024, 256, 0, stream>>>(x_var, NC, stats);
        ln_apply<<<lnGrid, 256, 0, stream>>>(x_var, NC, CDIM, stats,
                                             norm_w + (L * 2 + 1) * CDIM, norm_b + (L * 2 + 1) * CDIM);
        // kqv
        gemm(x_op, 0, CDIM, nullptr, 0, convkqvT + (size_t)(L * 2 + 0) * 768 * CDIM, 0, CDIM,
             conv_kqv_b + (L * 2 + 0) * 768, z_op, 0, 768, nullptr, nullptr, nullptr,
             NOPN, 768, CDIM, 0, 1);
        gemm(x_var, 0, CDIM, nullptr, 0, convkqvT + (size_t)(L * 2 + 1) * 768 * CDIM, 0, CDIM,
             conv_kqv_b + (L * 2 + 1) * 768, z_var, 0, 768, nullptr, nullptr, nullptr,
             NVARN, 768, CDIM, 0, 1);
        // softmax state init
        fill(m_op, (long)NOPN * HN, NEG_INF);  fill(s_op, (long)NOPN * HN, 0.f);
        fill(m_var, (long)NVARN * HN, NEG_INF); fill(s_var, (long)NVARN * HN, 0.f);
        fill(agg_op, NC, 0.f); fill(agg_var, NC, 0.f);

        // per edge-type: K' transform (per-src-node, WMMA, batched heads) + alpha + max
        for (int t = 0; t < 3; ++t) {
            const float* zsrc = (t == 2) ? z_var : z_op;
            const float* qz = (t == 1) ? z_var : z_op;
            float* mB = (t == 1) ? m_var : m_op;
            gemm(zsrc + 0, 64, 768, nullptr, 0,
                 convkrT + (size_t)(L * 3 + t) * HN * DN * DN, DN * DN, DN, nullptr,
                 Rp, 64, CDIM, nullptr, nullptr, nullptr, NOPN, DN, DN, 0, HN);
            edge_alpha<<<eblk(ECD), 256, 0, stream>>>(qz + 256, 768, Rp, srcs[t], dsts[t], ECD,
                                                      conv_pr + (L * 3 + t) * HN, alphab[t]);
            seg_max_k<<<eblk(ECD), 256, 0, stream>>>(alphab[t], dsts[t], ECD, mB);
        }
        // exp + sum (max complete across all contributing types first)
        for (int t = 0; t < 3; ++t) {
            float* mB = (t == 1) ? m_var : m_op;
            float* sB = (t == 1) ? s_var : s_op;
            seg_expsum_k<<<eblk(ECD), 256, 0, stream>>>(alphab[t], dsts[t], ECD, mB, sB);
        }
        // V' transform + weighted aggregation
        for (int t = 0; t < 3; ++t) {
            const float* zsrc = (t == 2) ? z_var : z_op;
            float* sB = (t == 1) ? s_var : s_op;
            float* aggB = (t == 1) ? agg_var : agg_op;
            gemm(zsrc + 512, 64, 768, nullptr, 0,
                 convvrT + (size_t)(L * 3 + t) * HN * DN * DN, DN * DN, DN, nullptr,
                 Rp, 64, CDIM, nullptr, nullptr, nullptr, NOPN, DN, DN, 0, HN);
            seg_agg<<<eblk(ECD), 256, 0, stream>>>(alphab[t], srcs[t], dsts[t], ECD, sB, Rp, aggB);
        }
        // node_out: GELU(agg) @ W + b, sigmoid-skip residual, in place on x
        gemm(agg_op, 0, CDIM, nullptr, 1, convoutT + (size_t)(L * 2 + 0) * CDIM * CDIM, 0, CDIM,
             conv_out_b + (L * 2 + 0) * CDIM, x_op, 0, CDIM, nullptr, x_op,
             conv_skip + (L * 2 + 0), NOPN, CDIM, CDIM, 2, 1);
        gemm(agg_var, 0, CDIM, nullptr, 1, convoutT + (size_t)(L * 2 + 1) * CDIM * CDIM, 0, CDIM,
             conv_out_b + (L * 2 + 1) * CDIM, x_var, 0, CDIM, nullptr, x_var,
             conv_skip + (L * 2 + 1), NVARN, CDIM, CDIM, 2, 1);
    }

    // -------- pooling + head --------
    fill(padd_op, 16 * CDIM, 0.f); fill(pmax_op, 16 * CDIM, NEG_INF);
    fill(padd_var, 16 * CDIM, 0.f); fill(pmax_var, 16 * CDIM, NEG_INF);
    pool_accum<<<dim3((NOPN + 255) / 256), 256, 0, stream>>>(x_op, batch_op, NOPN, padd_op, pmax_op);
    pool_accum<<<dim3((NVARN + 255) / 256), 256, 0, stream>>>(x_var, batch_var, NVARN, padd_var, pmax_var);
    yb_mlp<<<1, 256, 0, stream>>>(y_base, yb_W1, yb_b1, yb_W2, yb_b2, ybuf);
    assemble_g<<<dim3((16 * 1040 + 255) / 256), 256, 0, stream>>>(padd_op, pmax_op, padd_var,
                                                                  pmax_var, ybuf, gbuf);
    final_mlp<<<16, 256, 0, stream>>>(gbuf, g_W1, g_b1, bn1_g, bn1_b, bn1_rm, bn1_rv,
                                      g_W2, g_b2, bn2_g, bn2_b, bn2_rm, bn2_rv,
                                      g_W3, g_b3, (float*)d_out);
}
